// GATBinaryGNN_18193481466581
// MI455X (gfx1250) — matrix operand order repaired and checked by reading, not compiled, not run on hardware
//
#include <hip/hip_runtime.h>
#include <hip/hip_bf16.h>

// ---------------------------------------------------------------------------
// GATBinaryGNN on gfx1250 (MI455X).
//  - All dense transforms on the matrix pipe: v_wmma_f32_16x16x32_bf16 with
//    operands pre-staged in bf16 (A zero-padded to Kp%32==0, W pre-packed into
//    the per-lane WMMA B-fragment layout) so the inner loop is pure
//    b128-load + wmma, no guards, no per-load waits.
//  - Edge softmax / scatter-add stays fp32 (bandwidth-bound, L2 atomics).
// ---------------------------------------------------------------------------

#define NL_   40000
#define NP_   40000
#define B_    64
#define H_    4
#define D_    64
#define NF_   256            // H*D
#define E_LL_ 160000
#define E_LP_ 240000
#define E_PL_ 240000

typedef __attribute__((ext_vector_type(16))) __bf16 v16bf;
typedef __attribute__((ext_vector_type(8)))  __bf16 v8bf;
typedef __attribute__((ext_vector_type(8)))  float  v8f;

// ---------------- utility fills ----------------
__global__ void fill_f32_kernel(float* p, float v, long n) {
  long t = (long)blockIdx.x * blockDim.x + threadIdx.x;
  if (t < n) p[t] = v;
}
__global__ void fill_u32_kernel(unsigned* p, unsigned v, long n) {
  long t = (long)blockIdx.x * blockDim.x + threadIdx.x;
  if (t < n) p[t] = v;
}

// ---------------- ordered-float atomic max helpers ----------------
__device__ inline unsigned f2ord(float f) {
  unsigned u = __float_as_uint(f);
  return (u & 0x80000000u) ? ~u : (u | 0x80000000u);
}
__device__ inline float ord2f(unsigned u) {
  return (u & 0x80000000u) ? __uint_as_float(u & 0x7fffffffu)
                           : __uint_as_float(~u);
}

// ---------------- fp32 -> bf16 row conversion with zero pad to Kp ----------
// grid: (M, ceil(Kp/256)), block: 256
__global__ void cvt_bf16_kernel(const float* __restrict__ src, __bf16* __restrict__ dst,
                                int K, int Kp)
{
  const int m = blockIdx.x;
  const int k = blockIdx.y * blockDim.x + threadIdx.x;
  if (k >= Kp) return;
  dst[(size_t)m * Kp + k] = (__bf16)((k < K) ? src[(size_t)m * K + k] : 0.f);
}

// ---------------- pack W[K,N] into per-lane WMMA B fragments ---------------
// Fragment layout (ISA 7.12.2, 16-bit B 32x16): lane L holds col (L&15),
// K = k0 + (L>=16 ? 8 : 0) + (i<8 ? i : i+8) for i in 0..15, 16 bf16 per lane.
// dst index: (((tn*KT + kk)*32 + lane)*16 + i)
__global__ void pack_b_kernel(const float* __restrict__ W, __bf16* __restrict__ dst,
                              int K, int Kp, int N)
{
  const int KT = Kp >> 5;
  const int t = blockIdx.x * blockDim.x + threadIdx.x;
  const int total = (N >> 4) * KT * 32;
  if (t >= total) return;
  const int lane = t & 31;
  const int kk = (t >> 5) % KT;
  const int tn = (t >> 5) / KT;
  const int half = lane >> 4, l15 = lane & 15;
  const int kbase = kk * 32 + half * 8;
  __bf16* o = dst + (size_t)t * 16;
#pragma unroll
  for (int i = 0; i < 16; ++i) {
    const int k = kbase + (i < 8 ? i : i + 8);
    o[i] = (__bf16)((k < K) ? W[(size_t)k * N + tn * 16 + l15] : 0.f);
  }
}

// ---------------- WMMA GEMM: C[M,N] = A[M,Kp]bf16 @ Bpacked (+bias, relu) --
// One wave per 16x16 C tile; Kp % 32 == 0, M % 16 == 0, N % 16 == 0.
// Hot loop: 2 aligned b128 loads (A frag) + 2 b128 loads (B frag) + 1 wmma.
__global__ __launch_bounds__(256)
void wmma_gemm_kernel(const __bf16* __restrict__ A, const __bf16* __restrict__ Bp,
                      const float* __restrict__ bias, float* __restrict__ C,
                      int M, int Kp, int N, int do_relu)
{
  const int lane = threadIdx.x;                            // 0..31
  const int tn   = blockIdx.x;                             // tile along N
  const int tm   = blockIdx.y * blockDim.y + threadIdx.y;  // tile along M
  if (tm * 16 >= M) return;

  const int half = lane >> 4;
  const int l15  = lane & 15;
  const int KT   = Kp >> 5;

  const __bf16* arow = A + (size_t)(tm * 16 + l15) * Kp + half * 8;
  const __bf16* bfr  = Bp + ((size_t)(tn * KT) * 32 + lane) * 16;

  v8f acc = {};
#pragma unroll 2
  for (int kk = 0; kk < KT; ++kk) {
    // A fragment: lane needs K = ka..ka+7 and ka+16..ka+23 (ka = kk*32+half*8)
    const v8bf lo = *(const v8bf*)(arow + (size_t)kk * 32);
    const v8bf hi = *(const v8bf*)(arow + (size_t)kk * 32 + 16);
    v16bf a;
#pragma unroll
    for (int i = 0; i < 8; ++i) { a[i] = lo[i]; a[i + 8] = hi[i]; }
    // B fragment: 16 contiguous bf16 per lane (pre-packed)
    const v16bf b = *(const v16bf*)(bfr + (size_t)kk * 512);
    acc = __builtin_amdgcn_wmma_f32_16x16x32_bf16(
        false, a, false, b, (short)0, acc, false, false);
  }

  // D layout: VGPR v -> row (v + 8*half), col = l15
  const int bcol = tn * 16 + l15;
#pragma unroll
  for (int v = 0; v < 8; ++v) {
    const long row = (long)tm * 16 + v + half * 8;
    float o = acc[v];
    if (bias) o += bias[bcol];
    if (do_relu) o = fmaxf(o, 0.f);
    C[row * N + bcol] = o;
  }
}

// ---------------- attention score: el[n,h] = sum_d feat[n,h,d]*avec[h,d] ---
__global__ void attn_score_kernel(const float* __restrict__ feat,
                                  const float* __restrict__ avec,
                                  float* __restrict__ out, int n)
{
  int t = blockIdx.x * blockDim.x + threadIdx.x;   // t = node*H + h
  if (t >= n * H_) return;
  int node = t >> 2, h = t & 3;
  const float* f = feat + (long)node * NF_ + h * D_;
  const float* a = avec + h * D_;
  float s = 0.f;
#pragma unroll 8
  for (int d = 0; d < D_; ++d) s += f[d] * a[d];
  out[t] = s;
}

// ---------------- edge pass 1: leaky_relu score + segment max --------------
__global__ void edge_max_kernel(const int* __restrict__ src, const int* __restrict__ dst,
                                const float* __restrict__ el, const float* __restrict__ er,
                                float* __restrict__ ebuf, unsigned* __restrict__ mord, int E)
{
  int t = blockIdx.x * blockDim.x + threadIdx.x;
  if (t >= E * H_) return;
  int e = t >> 2, h = t & 3;
  int s = src[e], d = dst[e];
  float x = el[s * H_ + h] + er[d * H_ + h];
  x = (x > 0.f) ? x : 0.2f * x;                 // leaky_relu(0.2)
  ebuf[t] = x;
  atomicMax(&mord[d * H_ + h], f2ord(x));
}

// ---------------- edge pass 2: ex = exp(e - m[dst]); segment sum -----------
__global__ void edge_expsum_kernel(const int* __restrict__ dst,
                                   float* __restrict__ ebuf,
                                   const unsigned* __restrict__ mord,
                                   float* __restrict__ ssum, int E)
{
  int t = blockIdx.x * blockDim.x + threadIdx.x;
  if (t >= E * H_) return;
  int e = t >> 2, h = t & 3;
  int d = dst[e];
  float ex = __expf(ebuf[t] - ord2f(mord[d * H_ + h]));
  ebuf[t] = ex;
  atomicAdd(&ssum[d * H_ + h], ex);
}

// ---------------- edge pass 3: acc[dst] += fs[src] * (ex / s[dst]) ---------
__global__ __launch_bounds__(256)
void edge_scatter_kernel(const int* __restrict__ src, const int* __restrict__ dst,
                         const float* __restrict__ ebuf, const float* __restrict__ ssum,
                         const float* __restrict__ fsrc, float* __restrict__ acc)
{
  const int e = blockIdx.x;
  const int c = threadIdx.x;               // 0..255
  const int h = c >> 6;
  const int s = src[e], d = dst[e];
  const float a = ebuf[e * H_ + h] / ssum[d * H_ + h];
  atomicAdd(&acc[(long)d * NF_ + c], fsrc[(long)s * NF_ + c] * a);
}

// ---------------- fused (acc + b1 [+ b2]) -> relu, in place ----------------
__global__ void bias_relu_kernel(float* __restrict__ acc, const float* __restrict__ b1,
                                 const float* __restrict__ b2, long n)
{
  long t = (long)blockIdx.x * blockDim.x + threadIdx.x;
  if (t >= n) return;
  int c = (int)(t & (NF_ - 1));
  float v = acc[t] + b1[c];
  if (b2) v += b2[c];
  acc[t] = fmaxf(v, 0.f);
}

// ---------------- per-graph mean readout -----------------------------------
__global__ __launch_bounds__(256)
void readout_sum_kernel(const float* __restrict__ h, const int* __restrict__ gid,
                        float* __restrict__ sums, float* __restrict__ counts, int colOff)
{
  const int node = blockIdx.x;
  const int c = threadIdx.x;
  const int g = gid[node];
  atomicAdd(&sums[(long)g * 512 + colOff + c], h[(long)node * NF_ + c]);
  if (c == 0) atomicAdd(&counts[g], 1.f);
}

__global__ void readout_mean_kernel(const float* __restrict__ sums,
                                    const float* __restrict__ cl,
                                    const float* __restrict__ cp,
                                    float* __restrict__ g)
{
  int t = blockIdx.x * blockDim.x + threadIdx.x;   // B*512
  if (t >= B_ * 512) return;
  int b = t >> 9, c = t & 511;
  float cnt = (c < 256) ? cl[b] : cp[b];
  g[t] = sums[t] / fmaxf(cnt, 1.f);
}

// ---------------------------------------------------------------------------
// Host-side orchestration
// ---------------------------------------------------------------------------
static inline float* carve_f(char*& p, size_t nfl) {
  float* r = (float*)p;
  p += ((nfl * sizeof(float) + 255) / 256) * 256;
  return r;
}
static inline unsigned* carve_u(char*& p, size_t nu) {
  unsigned* r = (unsigned*)p;
  p += ((nu * sizeof(unsigned) + 255) / 256) * 256;
  return r;
}
static inline __bf16* carve_bf(char*& p, size_t nel) {
  __bf16* r = (__bf16*)p;
  p += ((nel * 2 + 255) / 256) * 256;
  return r;
}

static void launch_cvt(hipStream_t s, const float* src, __bf16* dst, int M, int K, int Kp) {
  dim3 grd(M, (Kp + 255) / 256);
  cvt_bf16_kernel<<<grd, dim3(256), 0, s>>>(src, dst, K, Kp);
}
static void launch_pack(hipStream_t s, const float* W, __bf16* dst, int K, int Kp, int N) {
  int total = (N >> 4) * (Kp >> 5) * 32;
  pack_b_kernel<<<dim3((total + 255) / 256), dim3(256), 0, s>>>(W, dst, K, Kp, N);
}
static void launch_gemm(hipStream_t s, const __bf16* A, const __bf16* Bp, const float* bias,
                        float* C, int M, int Kp, int N, int relu)
{
  dim3 blk(32, 8);
  dim3 grd(N / 16, (M / 16 + 7) / 8);
  wmma_gemm_kernel<<<grd, blk, 0, s>>>(A, Bp, bias, C, M, Kp, N, relu);
}
// pack + gemm from fp32 W
static void gemm_w(hipStream_t s, const __bf16* Abf, const float* W, __bf16* wpack,
                   const float* bias, float* C, int M, int K, int Kp, int N, int relu)
{
  launch_pack(s, W, wpack, K, Kp, N);
  launch_gemm(s, Abf, wpack, bias, C, M, Kp, N, relu);
}

static void launch_fill_f(hipStream_t s, float* p, float v, long n) {
  fill_f32_kernel<<<dim3((unsigned)((n + 255) / 256)), dim3(256), 0, s>>>(p, v, n);
}
static void launch_fill_u(hipStream_t s, unsigned* p, unsigned v, long n) {
  fill_u32_kernel<<<dim3((unsigned)((n + 255) / 256)), dim3(256), 0, s>>>(p, v, n);
}

// One GAT relation: attention scores + 3-pass edge softmax + scatter into acc.
static void gat_relation(hipStream_t s,
                         const int* src, const int* dst, int E, int Nsrc, int Ndst,
                         const float* fsrc, const float* fdst,
                         const float* al, const float* ar,
                         float* acc,
                         float* el, float* er, float* ebuf, unsigned* mord, float* ssum)
{
  attn_score_kernel<<<dim3((Nsrc * H_ + 255) / 256), dim3(256), 0, s>>>(fsrc, al, el, Nsrc);
  attn_score_kernel<<<dim3((Ndst * H_ + 255) / 256), dim3(256), 0, s>>>(fdst, ar, er, Ndst);
  launch_fill_u(s, mord, 0u, (long)Ndst * H_);
  launch_fill_f(s, ssum, 0.f, (long)Ndst * H_);
  int eh = E * H_;
  edge_max_kernel<<<dim3((eh + 255) / 256), dim3(256), 0, s>>>(src, dst, el, er, ebuf, mord, E);
  edge_expsum_kernel<<<dim3((eh + 255) / 256), dim3(256), 0, s>>>(dst, ebuf, mord, ssum, E);
  edge_scatter_kernel<<<dim3(E), dim3(256), 0, s>>>(src, dst, ebuf, ssum, fsrc, acc);
}

extern "C" void kernel_launch(void* const* d_in, const int* in_sizes, int n_in,
                              void* d_out, int out_size, void* d_ws, size_t ws_size,
                              hipStream_t stream)
{
  (void)in_sizes; (void)n_in; (void)out_size; (void)ws_size;

  // ---- inputs (setup_inputs order) ----
  int i = 0;
  const float* x_lig  = (const float*)d_in[i++];   // [NL,74]
  const float* x_pro  = (const float*)d_in[i++];   // [NP,128]
  const float* W1_ll  = (const float*)d_in[i++];
  const float* al1_ll = (const float*)d_in[i++];
  const float* ar1_ll = (const float*)d_in[i++];
  const float* b1_ll  = (const float*)d_in[i++];
  const float* W1_lps = (const float*)d_in[i++];
  const float* W1_lpd = (const float*)d_in[i++];
  const float* al1_lp = (const float*)d_in[i++];
  const float* ar1_lp = (const float*)d_in[i++];
  const float* b1_lp  = (const float*)d_in[i++];
  const float* W1_pls = (const float*)d_in[i++];
  const float* W1_pld = (const float*)d_in[i++];
  const float* al1_pl = (const float*)d_in[i++];
  const float* ar1_pl = (const float*)d_in[i++];
  const float* b1_pl  = (const float*)d_in[i++];
  const float* W2_ll  = (const float*)d_in[i++];
  const float* al2_ll = (const float*)d_in[i++];
  const float* ar2_ll = (const float*)d_in[i++];
  const float* b2_ll  = (const float*)d_in[i++];
  const float* W2_lps = (const float*)d_in[i++];
  const float* W2_lpd = (const float*)d_in[i++];
  const float* al2_lp = (const float*)d_in[i++];
  const float* ar2_lp = (const float*)d_in[i++];
  const float* b2_lp  = (const float*)d_in[i++];
  const float* W2_pls = (const float*)d_in[i++];
  const float* W2_pld = (const float*)d_in[i++];
  const float* al2_pl = (const float*)d_in[i++];
  const float* ar2_pl = (const float*)d_in[i++];
  const float* b2_pl  = (const float*)d_in[i++];
  const float* W_mlp  = (const float*)d_in[i++];   // [512,256]
  const float* b_mlp  = (const float*)d_in[i++];   // [256]
  const int* src_ll   = (const int*)d_in[i++];
  const int* dst_ll   = (const int*)d_in[i++];
  const int* src_lp   = (const int*)d_in[i++];
  const int* dst_lp   = (const int*)d_in[i++];
  const int* src_pl   = (const int*)d_in[i++];
  const int* dst_pl   = (const int*)d_in[i++];
  const int* gid_lig  = (const int*)d_in[i++];
  const int* gid_pro  = (const int*)d_in[i++];

  // ---- workspace carve ----
  char* p = (char*)d_ws;
  const size_t FEAT = (size_t)NL_ * NF_;           // NL == NP == 40000
  float* accl1 = carve_f(p, FEAT);                 // layer-1 lig acc -> h_l
  float* accp1 = carve_f(p, FEAT);                 // layer-1 pro acc -> h_p
  float* accl2 = carve_f(p, FEAT);                 // layer-2 lig acc -> h_l2
  float* accp2 = carve_f(p, FEAT);                 // layer-2 pro acc -> h_p2
  float* ftS   = carve_f(p, FEAT);                 // per-relation src transform
  float* ftD   = carve_f(p, FEAT);                 // per-relation dst transform
  __bf16* xlig_bf = carve_bf(p, (size_t)NL_ * 96);    // x_lig  bf16, Kp=96
  __bf16* xpro_bf = carve_bf(p, (size_t)NP_ * 128);   // x_pro  bf16, Kp=128
  __bf16* hl_bf   = carve_bf(p, FEAT);                // h_l    bf16, Kp=256
  __bf16* hp_bf   = carve_bf(p, FEAT);                // h_p    bf16, Kp=256
  __bf16* g_bf    = carve_bf(p, (size_t)B_ * 512);    // readout bf16, Kp=512
  __bf16* wpack   = carve_bf(p, (size_t)512 * 256);   // packed W (max Kp*N)
  float* el    = carve_f(p, (size_t)NL_ * H_);
  float* er    = carve_f(p, (size_t)NL_ * H_);
  float* ebuf  = carve_f(p, (size_t)E_LP_ * H_);   // max E
  unsigned* mord = carve_u(p, (size_t)NL_ * H_);
  float* ssum  = carve_f(p, (size_t)NL_ * H_);
  float* sums  = carve_f(p, (size_t)B_ * 512);
  float* cntl  = carve_f(p, B_);
  float* cntp  = carve_f(p, B_);
  float* gmat  = carve_f(p, (size_t)B_ * 512);

  hipStream_t s = stream;

  // ---- stage inputs to padded bf16 once ----
  launch_cvt(s, x_lig, xlig_bf, NL_, 74, 96);
  launch_cvt(s, x_pro, xpro_bf, NP_, 128, 128);

  // ===================== Layer 1 =====================
  launch_fill_f(s, accl1, 0.f, (long)FEAT);
  launch_fill_f(s, accp1, 0.f, (long)FEAT);

  // relation ll (lig->lig, shared W for src/dst)
  gemm_w(s, xlig_bf, W1_ll, wpack, nullptr, ftS, NL_, 74, 96, NF_, 0);
  gat_relation(s, src_ll, dst_ll, E_LL_, NL_, NL_, ftS, ftS, al1_ll, ar1_ll,
               accl1, el, er, ebuf, mord, ssum);

  // relation pl (pro->lig)
  gemm_w(s, xpro_bf, W1_pls, wpack, nullptr, ftS, NP_, 128, 128, NF_, 0);
  gemm_w(s, xlig_bf, W1_pld, wpack, nullptr, ftD, NL_, 74, 96, NF_, 0);
  gat_relation(s, src_pl, dst_pl, E_PL_, NP_, NL_, ftS, ftD, al1_pl, ar1_pl,
               accl1, el, er, ebuf, mord, ssum);

  // relation lp (lig->pro)
  gemm_w(s, xlig_bf, W1_lps, wpack, nullptr, ftS, NL_, 74, 96, NF_, 0);
  gemm_w(s, xpro_bf, W1_lpd, wpack, nullptr, ftD, NP_, 128, 128, NF_, 0);
  gat_relation(s, src_lp, dst_lp, E_LP_, NL_, NP_, ftS, ftD, al1_lp, ar1_lp,
               accp1, el, er, ebuf, mord, ssum);

  // h_l = relu(acc + b1_ll + b1_pl), h_p = relu(acc + b1_lp)   (in place)
  bias_relu_kernel<<<dim3((unsigned)((FEAT + 255) / 256)), dim3(256), 0, s>>>(accl1, b1_ll, b1_pl, (long)FEAT);
  bias_relu_kernel<<<dim3((unsigned)((FEAT + 255) / 256)), dim3(256), 0, s>>>(accp1, b1_lp, nullptr, (long)FEAT);

  // stage layer-1 outputs to bf16 once (used by 5 layer-2 GEMMs)
  launch_cvt(s, accl1, hl_bf, NL_, NF_, NF_);
  launch_cvt(s, accp1, hp_bf, NP_, NF_, NF_);

  // ===================== Layer 2 =====================
  launch_fill_f(s, accl2, 0.f, (long)FEAT);
  launch_fill_f(s, accp2, 0.f, (long)FEAT);

  // relation ll
  gemm_w(s, hl_bf, W2_ll, wpack, nullptr, ftS, NL_, NF_, NF_, NF_, 0);
  gat_relation(s, src_ll, dst_ll, E_LL_, NL_, NL_, ftS, ftS, al2_ll, ar2_ll,
               accl2, el, er, ebuf, mord, ssum);

  // relation pl
  gemm_w(s, hp_bf, W2_pls, wpack, nullptr, ftS, NP_, NF_, NF_, NF_, 0);
  gemm_w(s, hl_bf, W2_pld, wpack, nullptr, ftD, NL_, NF_, NF_, NF_, 0);
  gat_relation(s, src_pl, dst_pl, E_PL_, NP_, NL_, ftS, ftD, al2_pl, ar2_pl,
               accl2, el, er, ebuf, mord, ssum);

  // relation lp
  gemm_w(s, hl_bf, W2_lps, wpack, nullptr, ftS, NL_, NF_, NF_, NF_, 0);
  gemm_w(s, hp_bf, W2_lpd, wpack, nullptr, ftD, NP_, NF_, NF_, NF_, 0);
  gat_relation(s, src_lp, dst_lp, E_LP_, NL_, NP_, ftS, ftD, al2_lp, ar2_lp,
               accp2, el, er, ebuf, mord, ssum);

  bias_relu_kernel<<<dim3((unsigned)((FEAT + 255) / 256)), dim3(256), 0, s>>>(accl2, b2_ll, b2_pl, (long)FEAT);
  bias_relu_kernel<<<dim3((unsigned)((FEAT + 255) / 256)), dim3(256), 0, s>>>(accp2, b2_lp, nullptr, (long)FEAT);

  // ===================== Readout + MLP =====================
  launch_fill_f(s, sums, 0.f, (long)B_ * 512);
  launch_fill_f(s, cntl, 0.f, B_);
  launch_fill_f(s, cntp, 0.f, B_);
  readout_sum_kernel<<<dim3(NL_), dim3(256), 0, s>>>(accl2, gid_lig, sums, cntl, 0);
  readout_sum_kernel<<<dim3(NP_), dim3(256), 0, s>>>(accp2, gid_pro, sums, cntp, 256);
  readout_mean_kernel<<<dim3((B_ * 512 + 255) / 256), dim3(256), 0, s>>>(sums, cntl, cntp, gmat);

  // out = relu(g @ W_mlp + b_mlp)  -> [64,256]
  launch_cvt(s, gmat, g_bf, B_, 512, 512);
  gemm_w(s, g_bf, W_mlp, wpack, b_mlp, (float*)d_out, B_, 512, 512, NF_, 1);
}
